// NonLocalBlock_13030930776823
// MI455X (gfx1250) — matrix operand order repaired and checked by reading, not compiled, not run on hardware
//
#include <hip/hip_runtime.h>

typedef __attribute__((ext_vector_type(2)))  float  v2f;
typedef __attribute__((ext_vector_type(8)))  float  v8f;
typedef __attribute__((ext_vector_type(16))) __bf16 v16bf;

// Problem constants: B=16, C=64, H=W=64, HW=4096, C1=8, C2=32, pooled HW/4=1024

// ---------------------------------------------------------------------------
// Kernel 1a: delta = conv1x1(x, w_delta) -> stored as [B][4096][8]
// ---------------------------------------------------------------------------
__global__ __launch_bounds__(256)
void nlb_conv_delta(const float* __restrict__ x, const float* __restrict__ w,
                    const float* __restrict__ bias, float* __restrict__ out)
{
    int idx = blockIdx.x * 256 + threadIdx.x;   // 65536 = B*HW
    int b = idx >> 12, n = idx & 4095;
    float acc[8];
#pragma unroll
    for (int j = 0; j < 8; ++j) acc[j] = bias[j];
    const float* xp = x + (size_t)b * 64 * 4096 + n;
    for (int c = 0; c < 64; ++c) {
        float xv = xp[(size_t)c * 4096];
#pragma unroll
        for (int j = 0; j < 8; ++j) acc[j] += xv * w[j * 64 + c];
    }
    float* op = out + (size_t)idx * 8;
#pragma unroll
    for (int j = 0; j < 8; ++j) op[j] = acc[j];
}

// ---------------------------------------------------------------------------
// Kernel 1b: phi = maxpool2(conv1x1(x, w_phi)), with the source-faithful
// permute(1,0,2,3)-then-reshape batch/channel shuffle:
//   phi_used[i][j][m] = pooled[c1][bb][m], l=i*8+j, c1=l>>4, bb=l&15
// stored as [B][8][1024]
// ---------------------------------------------------------------------------
__global__ __launch_bounds__(256)
void nlb_conv_phi(const float* __restrict__ x, const float* __restrict__ w,
                  const float* __restrict__ bias, float* __restrict__ out)
{
    int idx = blockIdx.x * 256 + threadIdx.x;   // 131072 = 128 * 1024
    int l = idx >> 10, m = idx & 1023;
    int c1 = l >> 4, bb = l & 15;
    int ph = m >> 5, pw = m & 31;
    const float* xp = x + (size_t)bb * 64 * 4096;
    float best = -3.0e38f;
#pragma unroll
    for (int dy = 0; dy < 2; ++dy)
#pragma unroll
        for (int dx = 0; dx < 2; ++dx) {
            int pos = (2 * ph + dy) * 64 + 2 * pw + dx;
            float acc = bias[c1];
            for (int c = 0; c < 64; ++c) acc += xp[(size_t)c * 4096 + pos] * w[c1 * 64 + c];
            best = fmaxf(best, acc);
        }
    out[(size_t)(l >> 3) * 8192 + (size_t)(l & 7) * 1024 + m] = best;
}

// ---------------------------------------------------------------------------
// Kernel 1c: g = maxpool2(conv1x1(x, w_g)) -> stored as [B][1024][32]
// ---------------------------------------------------------------------------
__global__ __launch_bounds__(256)
void nlb_conv_g(const float* __restrict__ x, const float* __restrict__ w,
                const float* __restrict__ bias, float* __restrict__ out)
{
    int idx = blockIdx.x * 256 + threadIdx.x;   // 524288 = 16*1024*32
    int b = idx >> 15, m = (idx >> 5) & 1023, k = idx & 31;
    int ph = m >> 5, pw = m & 31;
    const float* xp = x + (size_t)b * 64 * 4096;
    float best = -3.0e38f;
#pragma unroll
    for (int dy = 0; dy < 2; ++dy)
#pragma unroll
        for (int dx = 0; dx < 2; ++dx) {
            int pos = (2 * ph + dy) * 64 + 2 * pw + dx;
            float acc = bias[k];
            for (int c = 0; c < 64; ++c) acc += xp[(size_t)c * 4096 + pos] * w[k * 64 + c];
            best = fmaxf(best, acc);
        }
    out[(size_t)idx] = best;   // ((b*1024+m)*32+k) == idx
}

// ---------------------------------------------------------------------------
// Transposed 16x16 score tile for m-range [mb, mb+16):
//   T = phi^T(m x 8) @ delta^T(8 x n), via V_WMMA_F32_16X16X4_F32 (K=8 -> 2 ops)
//   C layout: lane holds n = lane&15, m_local = 8*(lane>>4) + v
// ---------------------------------------------------------------------------
__device__ __forceinline__
v8f nlb_score16(const float* __restrict__ phi_s, int mb, int j0, int lr,
                v2f bLo, v2f bHi)
{
    int mcol = mb + lr;
    v2f aLo = { phi_s[(j0 + 0) * 1024 + mcol], phi_s[(j0 + 1) * 1024 + mcol] };
    v2f aHi = { phi_s[(j0 + 4) * 1024 + mcol], phi_s[(j0 + 5) * 1024 + mcol] };
    v8f z = {};
    v8f T = __builtin_amdgcn_wmma_f32_16x16x4_f32(false, aLo, false, bLo, (short)0, z, false, false);
    T     = __builtin_amdgcn_wmma_f32_16x16x4_f32(false, aHi, false, bHi, (short)0, T, false, false);
    return T;
}

// ---------------------------------------------------------------------------
// Kernel 2: fused attention, transposed-score formulation, two-pass softmax.
//  grid = 16 batches * 32 row-tiles; block = 256 = 8 waves; 16 rows per wave.
//  phi staged to LDS via GLOBAL_LOAD_ASYNC_TO_LDS_B128 (ASYNCcnt), overlapped
//  with the gB swizzle. Pass 1: row max (per-lane VALU, 4 parallel chains).
//  Pass 2: exp -> in-register bf16 pack (transposed C-layout == bf16 A-layout)
//  -> V_WMMA_F32_16X16X32_BF16.
// ---------------------------------------------------------------------------
__global__ __launch_bounds__(256)
void nlb_attn(const float* __restrict__ delta, const float* __restrict__ phi,
              const float* __restrict__ g, float* __restrict__ ao)
{
    __shared__ float  phi_s[8 * 1024];          // 32 KB, [j][m]
    __shared__ __bf16 gB[32 * 2 * 32 * 16];     // 64 KB, WMMA-B-layout per chunk/tile

    const int b    = blockIdx.x >> 5;
    const int tile = blockIdx.x & 31;
    const int tid  = threadIdx.x;

    // ---- async DMA phi[b] -> phi_s: 2048 x 16B, 8 per lane, no VGPR round-trip
    {
        const char* gsrc = (const char*)(phi + (size_t)b * 8192);
        unsigned lbase = (unsigned)(size_t)(&phi_s[0]);
#pragma unroll
        for (int i = 0; i < 8; ++i) {
            unsigned off = (unsigned)((i * 256 + tid) * 16);
            asm volatile("global_load_async_to_lds_b128 %0, %1, off"
                         :: "v"(lbase + off), "v"(gsrc + off) : "memory");
        }
    }

    // ---- meanwhile: swizzle g[b] into exact bf16 B-matrix lane layout:
    //  value(m,k) -> gB[((c*2+t)*32+lane)*16+e], c=m>>5, e=(m&7)|((m&16)>>1), t=k>>4,
    //  lane=((m&8)<<1)|(k&15)   (inner loop: two contiguous b128 reads per tile)
    for (int s = tid; s < 1024 * 32; s += 256) {
        int m = s >> 5, k = s & 31;
        float val = g[((size_t)b * 1024 + m) * 32 + k];
        int c = m >> 5, within = m & 31;
        int lane_d = ((within & 8) << 1) | (k & 15);
        int t = k >> 4, e = (within & 7) | ((within & 16) >> 1);
        gB[(((c * 2 + t) * 32) + lane_d) * 16 + e] = (__bf16)val;
    }

    asm volatile("s_wait_asynccnt 0" ::: "memory");
    __syncthreads();

    const int wave = tid >> 5, lane = tid & 31;
    const int lr = lane & 15, half = lane >> 4;
    const int rb = tile * 128 + wave * 16;
    const int n  = rb + lr;                     // this lane's attention row
    const int kb = half * 2;                    // f32 B: K = (lane>>4)*2 + v
    const int j0 = half * 2;                    // f32 A: K = (lane>>4)*2 + v

    // delta^T is the loop-invariant B operand (4 regs per lane)
    const float* dp = delta + ((size_t)b * 4096 + n) * 8;
    v2f bLo = { dp[kb],     dp[kb + 1] };       // j = 0..3
    v2f bHi = { dp[4 + kb], dp[5 + kb] };       // j = 4..7

    // ---- Pass 1: global row max; 4 independent max chains ----
    float gm0 = -3.0e38f, gm1 = -3.0e38f, gm2 = -3.0e38f, gm3 = -3.0e38f;
    for (int c = 0; c < 32; ++c) {
        v8f T0 = nlb_score16(phi_s, c * 32,      j0, lr, bLo, bHi);
        v8f T1 = nlb_score16(phi_s, c * 32 + 16, j0, lr, bLo, bHi);
        gm0 = fmaxf(gm0, fmaxf(T0[0], T1[0]));
        gm1 = fmaxf(gm1, fmaxf(T0[1], T1[1]));
        gm2 = fmaxf(gm2, fmaxf(T0[2], T1[2]));
        gm3 = fmaxf(gm3, fmaxf(T0[3], T1[3]));
        gm0 = fmaxf(gm0, fmaxf(T0[4], T1[4]));
        gm1 = fmaxf(gm1, fmaxf(T0[5], T1[5]));
        gm2 = fmaxf(gm2, fmaxf(T0[6], T1[6]));
        gm3 = fmaxf(gm3, fmaxf(T0[7], T1[7]));
    }
    float gmax = fmaxf(fmaxf(gm0, gm1), fmaxf(gm2, gm3));
    gmax = fmaxf(gmax, __shfl_xor(gmax, 16));   // merge the two m-halves

    // ---- Pass 2: exp, P@g accumulate, row-sum (4 parallel sum chains) ----
    v8f acc0 = {}, acc1 = {};
    float ss0 = 0.0f, ss1 = 0.0f, ss2 = 0.0f, ss3 = 0.0f;
    for (int c = 0; c < 32; ++c) {
        v8f T0 = nlb_score16(phi_s, c * 32,      j0, lr, bLo, bHi);
        v8f T1 = nlb_score16(phi_s, c * 32 + 16, j0, lr, bLo, bHi);

        // Transposed C-layout (n=lane&15, m=8*half+v / +16) is EXACTLY the
        // bf16 A-matrix layout (M=lane&15, K=(lane>=16?8:0)+(e&7)+(e>>3)*16):
        // pack P in registers, no LDS staging needed.
        v16bf Av;
        float sums[4] = {0.f, 0.f, 0.f, 0.f};
#pragma unroll
        for (int v = 0; v < 8; ++v) {
            float p0 = __expf(T0[v] - gmax);
            float p1 = __expf(T1[v] - gmax);
            sums[v & 3] += p0 + p1;
            Av[v]     = (__bf16)p0;
            Av[8 + v] = (__bf16)p1;
        }
        ss0 += sums[0]; ss1 += sums[1]; ss2 += sums[2]; ss3 += sums[3];

        union { v16bf v; uint4 q[2]; } B0, B1;
        const __bf16* g0p = gB + (((c * 2 + 0) * 32) + lane) * 16;
        const __bf16* g1p = gB + (((c * 2 + 1) * 32) + lane) * 16;
        B0.q[0] = *(const uint4*)(g0p); B0.q[1] = *(const uint4*)(g0p + 8);
        B1.q[0] = *(const uint4*)(g1p); B1.q[1] = *(const uint4*)(g1p + 8);

        acc0 = __builtin_amdgcn_wmma_f32_16x16x32_bf16(false, Av, false, B0.v, (short)0, acc0, false, false);
        acc1 = __builtin_amdgcn_wmma_f32_16x16x32_bf16(false, Av, false, B1.v, (short)0, acc1, false, false);
    }
    float ssum = (ss0 + ss1) + (ss2 + ss3);
    ssum += __shfl_xor(ssum, 16);               // full row sum for n
    float linv = 1.0f / ssum;

    // ---- Normalize + store. acc C-layout: lane holds rows r=8*half+v, col lr.
    //      1/l for tile-row r lives at lane r (score layout). 8 one-time shfls.
#pragma unroll
    for (int v = 0; v < 8; ++v) {
        int r = half * 8 + v;
        float inv = __shfl(linv, r);
        size_t base = ((size_t)b * 4096 + rb + r) * 32;
        ao[base + lr]      = acc0[v] * inv;
        ao[base + 16 + lr] = acc1[v] * inv;
    }
}

// ---------------------------------------------------------------------------
// Kernel 3: out = gamma * conv1x1(ao, w_last) + x
// ---------------------------------------------------------------------------
__global__ __launch_bounds__(256)
void nlb_outproj(const float* __restrict__ x, const float* __restrict__ ao,
                 const float* __restrict__ w_last, const float* __restrict__ b_last,
                 const float* __restrict__ gamma, float* __restrict__ out)
{
    __shared__ float aos[128 * 33];   // padded stride -> no bank conflicts
    __shared__ float wl[2048];
    __shared__ float bl[64];
    int b  = blockIdx.x >> 5;
    int n0 = (blockIdx.x & 31) * 128;
    int tid = threadIdx.x;
    for (int s = tid; s < 4096; s += 256) {
        int np = s >> 5, k = s & 31;
        aos[np * 33 + k] = ao[((size_t)b * 4096 + n0 + np) * 32 + k];
    }
    for (int s = tid; s < 2048; s += 256) wl[s] = w_last[s];
    if (tid < 64) bl[tid] = b_last[tid];
    __syncthreads();
    float gm = gamma[0];
    int np = tid & 127, cc0 = tid >> 7;
    int n = n0 + np;
    for (int cc = cc0; cc < 64; cc += 2) {
        float acc = bl[cc];
#pragma unroll
        for (int k = 0; k < 32; ++k) acc += wl[cc * 32 + k] * aos[np * 33 + k];
        size_t oi = ((size_t)b * 64 + cc) * 4096 + n;
        out[oi] = gm * acc + x[oi];
    }
}

// ---------------------------------------------------------------------------
extern "C" void kernel_launch(void* const* d_in, const int* in_sizes, int n_in,
                              void* d_out, int out_size, void* d_ws, size_t ws_size,
                              hipStream_t stream) {
    const float* x       = (const float*)d_in[0];
    const float* w_delta = (const float*)d_in[1];
    const float* b_delta = (const float*)d_in[2];
    const float* w_phi   = (const float*)d_in[3];
    const float* b_phi   = (const float*)d_in[4];
    const float* w_g     = (const float*)d_in[5];
    const float* b_g     = (const float*)d_in[6];
    const float* w_last  = (const float*)d_in[7];
    const float* b_last  = (const float*)d_in[8];
    const float* gamma   = (const float*)d_in[9];
    float* out = (float*)d_out;

    float* wsf   = (float*)d_ws;
    float* delta = wsf;                 // 16*4096*8   = 524288 f
    float* phi   = wsf + 524288;        // 16*8*1024   = 131072 f
    float* g     = wsf + 655360;        // 16*1024*32  = 524288 f
    float* ao    = wsf + 1179648;       // 16*4096*32  = 2097152 f  (13.1 MB total)

    nlb_conv_delta<<<256,  256, 0, stream>>>(x, w_delta, b_delta, delta);
    nlb_conv_phi  <<<512,  256, 0, stream>>>(x, w_phi,   b_phi,   phi);
    nlb_conv_g    <<<2048, 256, 0, stream>>>(x, w_g,     b_g,     g);
    nlb_attn      <<<512,  256, 0, stream>>>(delta, phi, g, ao);
    nlb_outproj   <<<512,  256, 0, stream>>>(x, ao, w_last, b_last, gamma, out);
}
